// GlobalAddAttention_57346403336500
// MI455X (gfx1250) — matrix operand order repaired and checked by reading, not compiled, not run on hardware
//
#include <hip/hip_runtime.h>

// ---------------------------------------------------------------------------
// CDNA5 (gfx1250) bf16 WMMA implementation of GlobalAddAttention.
// E=128, H=4, HD=32 hardcoded (matches reference). wave32, WMMA 16x16x32 bf16.
// ---------------------------------------------------------------------------

typedef __attribute__((ext_vector_type(16))) __bf16 v16bf;
typedef __attribute__((ext_vector_type(8)))  float  v8f;

union FragB { v16bf v; unsigned u[8]; };

#define LCAP 224   // max padded sequence length per graph (multiple of 32)

__device__ inline unsigned short f2bf(float x) {
  union { float f; unsigned u; } c; c.f = x;
  unsigned r = c.u + 0x7FFFu + ((c.u >> 16) & 1u);   // round-to-nearest-even
  return (unsigned short)(r >> 16);
}
__device__ inline unsigned pack2bf(float a, float b) {
  return (unsigned)f2bf(a) | ((unsigned)f2bf(b) << 16);
}
__device__ inline v8f wmma_bf16(v16bf a, v16bf b, v8f c) {
  return __builtin_amdgcn_wmma_f32_16x16x32_bf16(false, a, false, b,
                                                 (short)0, c, false, false);
}
// A-fragment (16x32 bf16) gathered from a row-major bf16 buffer packed as u32
// pairs. ISA layout: lane = 16*half + row; dword j holds K pair
// K0 = kbase + (j>=4 ? 16 : 0) + 8*half + 2*(j&3).
__device__ inline v16bf afrag(const unsigned* p, int rowStrideU32, int kbase, int lane) {
  int row = lane & 15, half = lane >> 4;
  FragB f;
#pragma unroll
  for (int j = 0; j < 8; ++j) {
    int K0 = kbase + ((j & 4) ? 16 : 0) + 8 * half + 2 * (j & 3);
    f.u[j] = p[row * rowStrideU32 + (K0 >> 1)];
  }
  return f.v;
}
__device__ inline v16bf bfrag_packed(const unsigned* p) {
  FragB f;
#pragma unroll
  for (int j = 0; j < 8; ++j) f.u[j] = p[j];
  return f.v;
}

// ------------------------------ setup kernels ------------------------------

__global__ void count_kernel(const int* __restrict__ batch, int* __restrict__ counts, int N) {
  int i = blockIdx.x * 256 + threadIdx.x;
  if (i < N) atomicAdd(&counts[batch[i]], 1);
}

__global__ void scan_kernel(const int* __restrict__ counts, int* __restrict__ offsets, int B) {
  __shared__ int sh[512];
  int tid = threadIdx.x;
  int v = (tid < B) ? counts[tid] : 0;
  sh[tid] = v;
  __syncthreads();
  for (int off = 1; off < 512; off <<= 1) {
    int t = (tid >= off) ? sh[tid - off] : 0;
    __syncthreads();
    sh[tid] += t;
    __syncthreads();
  }
  if (tid < B) offsets[tid] = sh[tid] - v;   // exclusive prefix sum
}

__global__ void pos_kernel(const int* __restrict__ batch, int* __restrict__ token_src, int N) {
  int i = blockIdx.x * 256 + threadIdx.x;
  if (i < N) token_src[i + batch[i] + 1] = i;   // node token slot
}

__global__ void metal_kernel(const int* __restrict__ offsets, int* __restrict__ token_src, int B) {
  int g = blockIdx.x * 256 + threadIdx.x;
  if (g < B) token_src[offsets[g] + g] = -(g + 1);   // metal token slot 0
}

// Pre-pack W (used as out = in @ W^T, i.e. B[k,n] = W[n,k]) into bf16 WMMA
// B-fragment order: wpack[mat*8192 + ((nt*4+kt)*32 + lane)*8 + j].
__global__ void pack_kernel(const float* __restrict__ Wk, const float* __restrict__ Wq,
                            const float* __restrict__ in_w, const float* __restrict__ out_w,
                            unsigned* __restrict__ wpack) {
  int idx = blockIdx.x * 256 + threadIdx.x;   // 0 .. 6*8192-1
  int mat = idx >> 13;
  int d = idx & 8191;
  int j = d & 7;
  int lane = (d >> 3) & 31;
  int fi = d >> 8;            // nt*4 + kt
  int kt = fi & 3, nt = fi >> 2;
  int n = nt * 16 + (lane & 15);
  int half = lane >> 4;
  int K0 = kt * 32 + ((j & 4) ? 16 : 0) + 8 * half + 2 * (j & 3);
  const float* W;
  switch (mat) {
    case 0: W = Wk; break;
    case 1: W = Wq; break;
    case 2: W = in_w; break;               // q in-proj
    case 3: W = in_w + 16384; break;       // k in-proj
    case 4: W = in_w + 32768; break;       // v in-proj
    default: W = out_w; break;
  }
  wpack[idx] = pack2bf(W[n * 128 + K0], W[n * 128 + K0 + 1]);
}

// ------------------------- stage 1: fused projections -----------------------
// One 16-token tile per block; 8 waves, wave w owns output n-tile w.
__global__ __launch_bounds__(256) void proj_kernel(
    const float* __restrict__ x, const float* __restrict__ metal_x,
    const int* __restrict__ token_src, const unsigned* __restrict__ wpack,
    const float* __restrict__ bk, const float* __restrict__ bq,
    const float* __restrict__ in_b,
    unsigned short* __restrict__ qg, unsigned short* __restrict__ kg,
    unsigned short* __restrict__ vg, int T) {
  __shared__ unsigned val_sh[16 * 64];
  __shared__ unsigned kin_sh[16 * 64];
  __shared__ unsigned qin_sh[16 * 64];
  const int tid = threadIdx.x;
  const int tile = blockIdx.x;
  {
    int row = tid >> 4;
    int c0 = (tid & 15) * 8;
    int t = tile * 16 + row;
    float v[8];
    if (t < T) {
      int src = token_src[t];
      const float* p = (src >= 0) ? (x + (size_t)src * 128)
                                  : (metal_x + (size_t)(-src - 1) * 128);
#pragma unroll
      for (int j = 0; j < 8; ++j) v[j] = p[c0 + j];
    } else {
#pragma unroll
      for (int j = 0; j < 8; ++j) v[j] = 0.f;
    }
#pragma unroll
    for (int j = 0; j < 4; ++j)
      val_sh[row * 64 + (c0 >> 1) + j] = pack2bf(v[2 * j], v[2 * j + 1]);
  }
  __syncthreads();
  const int lane = tid & 31;
  const int w = tid >> 5;
  const int half = lane >> 4;
  const int ncol = w * 16 + (lane & 15);

  // k_in = relu(value @ Wk^T + bk), q_in = relu(value @ Wq^T + bq)
  v8f acck = {}; v8f accq = {};
#pragma unroll
  for (int kt = 0; kt < 4; ++kt) {
    v16bf a = afrag(val_sh, 64, kt * 32, lane);
    acck = wmma_bf16(a, bfrag_packed(wpack + 0 * 8192 + ((w * 4 + kt) * 32 + lane) * 8), acck);
    accq = wmma_bf16(a, bfrag_packed(wpack + 1 * 8192 + ((w * 4 + kt) * 32 + lane) * 8), accq);
  }
  {
    float bkv = bk[ncol], bqv = bq[ncol];
    unsigned short* k16 = (unsigned short*)kin_sh;
    unsigned short* q16 = (unsigned short*)qin_sh;
#pragma unroll
    for (int r = 0; r < 8; ++r) {
      int m = r + 8 * half;
      float kv = acck[r] + bkv; kv = kv > 0.f ? kv : 0.f;
      float qv = accq[r] + bqv; qv = qv > 0.f ? qv : 0.f;
      k16[m * 128 + ncol] = f2bf(kv);
      q16[m * 128 + ncol] = f2bf(qv);
    }
  }
  __syncthreads();

  // q = q_in @ Wq_in^T, k = k_in @ Wk_in^T, v = value @ Wv_in^T (+bias)
  v8f aq = {}; v8f ak = {}; v8f av = {};
#pragma unroll
  for (int kt = 0; kt < 4; ++kt) {
    aq = wmma_bf16(afrag(qin_sh, 64, kt * 32, lane),
                   bfrag_packed(wpack + 2 * 8192 + ((w * 4 + kt) * 32 + lane) * 8), aq);
    ak = wmma_bf16(afrag(kin_sh, 64, kt * 32, lane),
                   bfrag_packed(wpack + 3 * 8192 + ((w * 4 + kt) * 32 + lane) * 8), ak);
    av = wmma_bf16(afrag(val_sh, 64, kt * 32, lane),
                   bfrag_packed(wpack + 4 * 8192 + ((w * 4 + kt) * 32 + lane) * 8), av);
  }
  float bq2 = in_b[ncol], bk2 = in_b[128 + ncol], bv2 = in_b[256 + ncol];
#pragma unroll
  for (int r = 0; r < 8; ++r) {
    int m = r + 8 * half;
    size_t t = (size_t)(tile * 16 + m);
    qg[t * 128 + ncol] = f2bf(aq[r] + bq2);
    kg[t * 128 + ncol] = f2bf(ak[r] + bk2);
    vg[t * 128 + ncol] = f2bf(av[r] + bv2);
  }
}

// ------------------------ stage 2: per-(graph,head) attention ---------------
// Block = 64 threads (2 waves); each wave processes alternating 16-query tiles.
__global__ __launch_bounds__(64) void attn_kernel(
    const unsigned short* __restrict__ qg, const unsigned short* __restrict__ kg,
    const unsigned short* __restrict__ vg, unsigned short* __restrict__ ctxg,
    const int* __restrict__ counts, const int* __restrict__ offsets, int T) {
  __shared__ unsigned Ksh[LCAP * 16];        // K: [key][32 bf16]      (14 KB)
  __shared__ unsigned VTshw[LCAP * 16];      // V^T: [d=32][LCAP bf16] (14 KB)
  __shared__ float Ssh[2 * 16 * LCAP];       // per-wave score rows    (28 KB)
  const int g = blockIdx.x >> 2;
  const int h = blockIdx.x & 3;
  int Lg = counts[g] + 1; if (Lg > LCAP) Lg = LCAP;
  const int base = offsets[g] + g;
  const int tid = threadIdx.x;

  const unsigned* kg32 = (const unsigned*)kg;
  for (int idx = tid; idx < Lg * 16; idx += 64)
    Ksh[idx] = kg32[(size_t)(base + (idx >> 4)) * 64 + h * 16 + (idx & 15)];

  unsigned short* VTsh = (unsigned short*)VTshw;
  const int Lpad = (Lg + 31) & ~31;
  for (int idx = tid; idx < 32 * Lpad; idx += 64) {
    int key = idx >> 5, dd = idx & 31;
    VTsh[dd * LCAP + key] =
        (key < Lg) ? vg[(size_t)(base + key) * 128 + h * 32 + dd] : (unsigned short)0;
  }
  __syncthreads();

  const int lane = tid & 31;
  const int w = tid >> 5;
  const int half = lane >> 4;
  const int nqt = (Lg + 15) >> 4;
  const int nkc = (Lg + 15) >> 4;
  const int nkc32 = (Lg + 31) >> 5;
  const unsigned* q32 = (const unsigned*)qg;
  const unsigned* VT32 = VTshw;
  float* Sw = Ssh + w * 16 * LCAP;
  unsigned* Sw32 = (unsigned*)Sw;
  const float scale = 0.1767766952966369f;   // 1/sqrt(32)

  for (int qt = w; qt < nqt; qt += 2) {
    FragB fq;   // Q A-fragment, 16 queries x 32 (head dim)
    {
      int tr = base + qt * 16 + (lane & 15);
      if (tr > T - 1) tr = T - 1;
#pragma unroll
      for (int j = 0; j < 8; ++j) {
        int K0 = ((j & 4) ? 16 : 0) + 8 * half + 2 * (j & 3);
        fq.u[j] = q32[(size_t)tr * 64 + h * 16 + (K0 >> 1)];
      }
    }
    // S = Q K^T * scale (masked)
    for (int kc = 0; kc < nkc; ++kc) {
      FragB fb;
      int nkey = kc * 16 + (lane & 15);
#pragma unroll
      for (int j = 0; j < 8; ++j) fb.u[j] = Ksh[nkey * 16 + ((j & 4) ? 8 : 0) + 4 * half + (j & 3)];
      v8f c = {};
      c = wmma_bf16(fq.v, fb.v, c);
#pragma unroll
      for (int r = 0; r < 8; ++r) {
        int m = r + 8 * half;
        Sw[m * LCAP + nkey] = (nkey < Lg) ? c[r] * scale : -3.0e38f;
      }
    }
    // row softmax, then pack P (bf16, normalized) in place; zero-pad to 32-mult
    if (lane < 16) {
      float* Sr = Sw + lane * LCAP;
      float mx = -3.0e38f;
      for (int c = 0; c < Lg; ++c) mx = fmaxf(mx, Sr[c]);
      float sum = 0.f;
      for (int c = 0; c < Lg; ++c) { float e = __expf(Sr[c] - mx); Sr[c] = e; sum += e; }
      float inv = 1.f / sum;
      unsigned* Pr = (unsigned*)Sr;
      int np = nkc32 * 32;
      for (int c = 0; c < np; c += 2) {
        float e0 = (c     < Lg) ? Sr[c] * inv     : 0.f;
        float e1 = (c + 1 < Lg) ? Sr[c + 1] * inv : 0.f;
        Pr[c >> 1] = pack2bf(e0, e1);
      }
    }
    // ctx = P V
#pragma unroll
    for (int nt = 0; nt < 2; ++nt) {
      v8f c2 = {};
      for (int kc = 0; kc < nkc32; ++kc) {
        v16bf a = afrag(Sw32, LCAP, kc * 32, lane);
        FragB fb;
        int dd = nt * 16 + (lane & 15);
#pragma unroll
        for (int j = 0; j < 8; ++j) {
          int K0 = kc * 32 + ((j & 4) ? 16 : 0) + 8 * half + 2 * (j & 3);
          fb.u[j] = VT32[dd * (LCAP / 2) + (K0 >> 1)];
        }
        c2 = wmma_bf16(a, fb.v, c2);
      }
#pragma unroll
      for (int r = 0; r < 8; ++r) {
        int m = r + 8 * half;
        int mg = qt * 16 + m;
        if (mg < Lg)
          ctxg[(size_t)(base + mg) * 128 + h * 32 + nt * 16 + (lane & 15)] = f2bf(c2[r]);
      }
    }
  }
}

// ----------------------- stage 3: out-proj + scatter ------------------------
__global__ __launch_bounds__(256) void out_kernel(
    const unsigned* __restrict__ ctx32, const unsigned* __restrict__ wpack,
    const float* __restrict__ out_b, const int* __restrict__ token_src,
    float* __restrict__ d_out, int T, int N) {
  const int tid = threadIdx.x, tile = blockIdx.x;
  const int lane = tid & 31, w = tid >> 5, half = lane >> 4;
  const int ncol = w * 16 + (lane & 15);
  v8f acc = {};
#pragma unroll
  for (int kt = 0; kt < 4; ++kt) {
    FragB fa;
    size_t t = (size_t)(tile * 16 + (lane & 15));
#pragma unroll
    for (int j = 0; j < 8; ++j) {
      int K0 = kt * 32 + ((j & 4) ? 16 : 0) + 8 * half + 2 * (j & 3);
      fa.u[j] = ctx32[t * 64 + (K0 >> 1)];
    }
    acc = wmma_bf16(fa.v, bfrag_packed(wpack + 5 * 8192 + ((w * 4 + kt) * 32 + lane) * 8), acc);
  }
  float bo = out_b[ncol];
#pragma unroll
  for (int r = 0; r < 8; ++r) {
    int m = r + 8 * half;
    int t = tile * 16 + m;
    if (t < T) {
      int src = token_src[t];
      float o = acc[r] + bo;
      if (src >= 0) d_out[(size_t)src * 128 + ncol] = o;                        // x_out
      else d_out[(size_t)N * 128 + (size_t)(-src - 1) * 128 + ncol] = o;        // metal_out
    }
  }
}

// ------------------------------- launcher -----------------------------------
extern "C" void kernel_launch(void* const* d_in, const int* in_sizes, int n_in,
                              void* d_out, int out_size, void* d_ws, size_t ws_size,
                              hipStream_t stream) {
  (void)n_in; (void)out_size; (void)ws_size;
  const float* x       = (const float*)d_in[0];
  const float* metal_x = (const float*)d_in[1];
  const int*   batch   = (const int*)d_in[2];
  const float* Wk      = (const float*)d_in[3];
  const float* bk      = (const float*)d_in[4];
  const float* Wq      = (const float*)d_in[5];
  const float* bq      = (const float*)d_in[6];
  const float* in_w    = (const float*)d_in[7];
  const float* in_b    = (const float*)d_in[8];
  const float* out_w   = (const float*)d_in[9];
  const float* out_b   = (const float*)d_in[10];

  const int N = in_sizes[0] / 128;
  const int B = in_sizes[1] / 128;
  const int T = N + B;
  const int ntiles = (T + 15) / 16;
  const int Tpad = ntiles * 16;

  char* ws = (char*)d_ws;
  size_t o = 0;
  auto carve = [&](size_t bytes) -> void* {
    void* p = ws + o;
    o = (o + bytes + 255) & ~(size_t)255;
    return p;
  };
  int* counts            = (int*)carve((size_t)B * 4);
  int* offsets           = (int*)carve((size_t)B * 4);
  int* token_src         = (int*)carve((size_t)Tpad * 4);
  unsigned* wpack        = (unsigned*)carve((size_t)6 * 8192 * 4);
  unsigned short* qg     = (unsigned short*)carve((size_t)Tpad * 128 * 2);
  unsigned short* kg     = (unsigned short*)carve((size_t)Tpad * 128 * 2);
  unsigned short* vg     = (unsigned short*)carve((size_t)Tpad * 128 * 2);
  unsigned short* ctxg   = (unsigned short*)carve((size_t)Tpad * 128 * 2);

  hipMemsetAsync(counts, 0, (size_t)B * 4, stream);
  count_kernel<<<(N + 255) / 256, 256, 0, stream>>>(batch, counts, N);
  scan_kernel<<<1, 512, 0, stream>>>(counts, offsets, B);
  pos_kernel<<<(N + 255) / 256, 256, 0, stream>>>(batch, token_src, N);
  metal_kernel<<<(B + 255) / 256, 256, 0, stream>>>(offsets, token_src, B);
  pack_kernel<<<192, 256, 0, stream>>>(Wk, Wq, in_w, out_w, wpack);
  proj_kernel<<<ntiles, 256, 0, stream>>>(x, metal_x, token_src, wpack,
                                          bk, bq, in_b, qg, kg, vg, T);
  attn_kernel<<<B * 4, 64, 0, stream>>>(qg, kg, vg, ctxg, counts, offsets, T);
  out_kernel<<<ntiles, 256, 0, stream>>>((const unsigned*)ctxg, wpack, out_b,
                                         token_src, (float*)d_out, T, N);
}
